// SingleGRUWaypoints_59416577573141
// MI455X (gfx1250) — matrix operand order repaired and checked by reading, not compiled, not run on hardware
//
#include <hip/hip_runtime.h>

typedef __attribute__((ext_vector_type(16))) __bf16 v16bf;
typedef __attribute__((ext_vector_type(8)))  float  v8f;
typedef __attribute__((ext_vector_type(4)))  float  f4;

// ---------------- CDNA5 data-mover availability probes -------------------
#if __has_include(<hip/amd_detail/amd_gfx1250_TDM.h>)
#define TDM_HDR 1
#else
#define TDM_HDR 0
#endif

#if defined(__has_builtin)
#if __has_builtin(__builtin_amdgcn_tensor_load_to_lds)
#if TDM_HDR
#define USE_TDM 6
#else
#define USE_TDM 5
#endif
#endif
#endif
#ifndef USE_TDM
#define USE_TDM 0
#endif

#if defined(__has_builtin) && __has_builtin(__builtin_amdgcn_global_load_async_to_lds_b128)
#define HAVE_ASYNC 1
#else
#define HAVE_ASYNC 0
#endif

#if USE_TDM == 6
#pragma message("probe: tensor_load_to_lds present, TDM header present -> using 6-arg form")
#elif USE_TDM == 5
#pragma message("probe: tensor_load_to_lds present, TDM header absent -> using 5-arg form")
#else
#pragma message("probe: tensor_load_to_lds builtin ABSENT")
#endif
#if HAVE_ASYNC
#pragma message("probe: global_load_async_to_lds_b128 builtin present")
#else
#pragma message("probe: global_load_async_to_lds_b128 builtin absent")
#endif
#if defined(__has_builtin) && __has_builtin(__builtin_amdgcn_s_wait_asynccnt)
#pragma message("probe: s_wait_asynccnt builtin present")
#else
#pragma message("probe: s_wait_asynccnt builtin absent")
#endif
#if defined(__has_builtin) && __has_builtin(__builtin_amdgcn_s_wait_tensorcnt)
#pragma message("probe: s_wait_tensorcnt builtin present")
#else
#pragma message("probe: s_wait_tensorcnt builtin absent")
#endif

// ---------------- LDS layout ----------------
// Phase 1 (head weights as bf16 hi/lo, padded rows for bank tiling)
#define HEAD_STRIDE   1040                       // bf16 elements per row (1024 + pad)
#define HEAD_HI_OFF   0
#define HEAD_LO_OFF   (64 * HEAD_STRIDE * 2)     // 133120 bytes
#define STAGE_OFF     (2 * 64 * HEAD_STRIDE * 2) // 266240: 2 x 16KB DMA staging
#define LDS_BYTES     (STAGE_OFF + 32768)        // 299008 bytes total (< 320KB)

// Phase 2 (overlaps phase-1 region; separated by barriers)
#define WHH_STRIDE    80                         // bf16 elements per row (64 + pad)
#define WHH_HI_OFF    0
#define WHH_LO_OFF    (192 * WHH_STRIDE * 2)     // 30720
#define BIH_OFF       61440                      // 192 f32
#define BHH_OFF       62208                      // 192 f32
#define WIH_OFF       62976                      // 384 f32
#define WOUT_OFF      64512                      // 128 f32
#define WOUTB_OFF     65024                      // 2 f32
#define X_OFF         65088                      // 128 rows * 2 f32
#define HS_OFF        66112                      // 8 waves * 16*68 f32
#define HS_STRIDE     68                         // f32 elements per row (64 + pad)
#define HS_WAVE_BYTES (16 * HS_STRIDE * 4)       // 4352

__device__ __forceinline__ v8f wmma_bf16(v16bf a, v16bf b, v8f c) {
  return __builtin_amdgcn_wmma_f32_16x16x32_bf16(false, a, false, b, (short)0, c,
                                                 false, false);
}
// 3-term split product: full = hi*hi + hi*lo + lo*hi  (error ~2^-19)
__device__ __forceinline__ v8f wmma3(v16bf ah, v16bf al, v16bf bh, v16bf bl, v8f c) {
  c = wmma_bf16(ah, bh, c);
  c = wmma_bf16(ah, bl, c);
  c = wmma_bf16(al, bh, c);
  return c;
}

__device__ __forceinline__ void splitA(const float* v, v16bf& hi, v16bf& lo) {
#pragma unroll
  for (int i = 0; i < 16; ++i) {
    float x = v[i];
    __bf16 hb = (__bf16)x;
    hi[i] = hb;
    lo[i] = (__bf16)(x - (float)hb);
  }
}

#if USE_TDM
typedef __attribute__((ext_vector_type(4))) unsigned int u32x4;
typedef __attribute__((ext_vector_type(8))) int          i32x8;
typedef __attribute__((ext_vector_type(4))) int          i32x4;

// TDM: DMA `rows` x `cols` fp32 tile (row-major, stride == cols) into LDS.
// D# per CDNA5 ISA ch.8: group0 = {count/flags, lds_addr, global_addr, type=2},
// group1 = {data_size=4B, tensor dims, tile dims, dim0 stride}.
__device__ __forceinline__ void tdm_load_rows_f32(const float* gsrc,
                                                  unsigned lds_off,
                                                  unsigned rows, unsigned cols) {
  unsigned long long ga = (unsigned long long)(size_t)gsrc;
  u32x4 g0;
  g0[0] = 1u;                                            // count=1 (valid, user)
  g0[1] = lds_off;                                       // lds_addr (bytes)
  g0[2] = (unsigned)(ga & 0xFFFFFFFFu);                  // global_addr[31:0]
  g0[3] = (unsigned)((ga >> 32) & 0x01FFFFFFu) | (2u << 30); // [56:32] | type=2
  i32x8 g1;
  g1[0] = (int)(2u << 16);                               // data_size code 2 = 4B
  g1[1] = (int)((cols & 0xFFFFu) << 16);                 // tensor_dim0[15:0]
  g1[2] = (int)((cols >> 16) & 0xFFFFu) |
          (int)((rows & 0xFFFFu) << 16);                 // dim0[31:16] | dim1[15:0]
  g1[3] = (int)((rows >> 16) & 0xFFFFu) |
          (int)((cols & 0xFFFFu) << 16);                 // dim1[31:16] | tile_dim0
  g1[4] = (int)(rows & 0xFFFFu);                         // tile_dim1 (tile_dim2=0)
  g1[5] = (int)cols;                                     // tensor_dim0_stride[31:0]
  g1[6] = 0;                                             // stride0[47:32]|stride1 lo
  g1[7] = 0;                                             // stride1 hi
  i32x4 z4 = {0, 0, 0, 0};
#if USE_TDM == 5
  __builtin_amdgcn_tensor_load_to_lds(g0, g1, z4, z4, 0);
#else
  i32x8 z8 = {0, 0, 0, 0, 0, 0, 0, 0};
  __builtin_amdgcn_tensor_load_to_lds(g0, g1, z4, z4, z8, 0);
#endif
}
#endif

#if !USE_TDM && HAVE_ASYNC
__device__ __forceinline__ void async_cp16(void* lds, const void* g) {
  __builtin_amdgcn_global_load_async_to_lds_b128(
      (__attribute__((address_space(1))) const void*)g,
      (__attribute__((address_space(3))) void*)lds, 0, 0);
}
#endif

__global__ void __launch_bounds__(256, 1)
gru_waypoints_kernel(const float* __restrict__ token,
                     const float* __restrict__ wp_head_w,
                     const float* __restrict__ wp_head_b,
                     const float* __restrict__ w_ih,
                     const float* __restrict__ w_hh,
                     const float* __restrict__ b_ih,
                     const float* __restrict__ b_hh,
                     const float* __restrict__ wp_out_w,
                     const float* __restrict__ wp_out_b,
                     const int*   __restrict__ nwp,
                     float*       __restrict__ out) {
  extern __shared__ __align__(32) char smem[];
  const int tid     = threadIdx.x;
  const int lane    = tid & 31;
  const int wave    = tid >> 5;
  const int nIdx    = lane & 15;        // col (B/C/D) or row (A) selector
  const int hiHalf  = lane >> 4;        // lane half selects K sub-range
  const int wgBase  = blockIdx.x * 128; // 8 waves * 16 rows
  const int T       = nwp[0];

  // ---------------- Stage phase-1 B matrix: wp_head_w -> LDS bf16 hi/lo ----
  {
    __bf16* hHi = (__bf16*)(smem + HEAD_HI_OFF);
    __bf16* hLo = (__bf16*)(smem + HEAD_LO_OFF);
#if USE_TDM || HAVE_ASYNC
    // Pipelined DMA: 16 chunks of 4 rows (16KB), double-buffered staging.
#if USE_TDM
    if (wave == 0) tdm_load_rows_f32(wp_head_w, STAGE_OFF, 4, 1024);
#else
    // all 256 threads: 4 x 16B async copies per chunk
    for (int i = 0; i < 4; ++i)
      async_cp16(smem + STAGE_OFF + (tid * 4 + i * 1024) * 4,
                 wp_head_w + (size_t)tid * 4 + (size_t)i * 1024);
#endif
#pragma unroll 1
    for (int c = 0; c < 16; ++c) {
#if USE_TDM
      if (wave == 0) {
        if (c + 1 < 16) {
          tdm_load_rows_f32(wp_head_w + (size_t)(c + 1) * 4096,
                            STAGE_OFF + (unsigned)(((c + 1) & 1) * 16384), 4, 1024);
          __builtin_amdgcn_s_wait_tensorcnt(1);   // in-order: chunk c complete
        } else {
          __builtin_amdgcn_s_wait_tensorcnt(0);
        }
      }
#else
      if (c + 1 < 16) {
        for (int i = 0; i < 4; ++i)
          async_cp16(smem + STAGE_OFF + ((c + 1) & 1) * 16384 + (tid * 4 + i * 1024) * 4,
                     wp_head_w + (size_t)(c + 1) * 4096 + (size_t)tid * 4 + (size_t)i * 1024);
        __builtin_amdgcn_s_wait_asynccnt(4);      // in-order: chunk c complete
      } else {
        __builtin_amdgcn_s_wait_asynccnt(0);
      }
#endif
      __syncthreads();                            // chunk c visible to all waves
      const float* st = (const float*)(smem + STAGE_OFF + (c & 1) * 16384);
      for (int e = tid * 4; e < 4096; e += 1024) {
        f4 v = *(const f4*)(st + e);
        int r = c * 4 + (e >> 10), k = e & 1023;
#pragma unroll
        for (int i = 0; i < 4; ++i) {
          float x = v[i];
          __bf16 hb = (__bf16)x;
          hHi[r * HEAD_STRIDE + k + i] = hb;
          hLo[r * HEAD_STRIDE + k + i] = (__bf16)(x - (float)hb);
        }
      }
      __syncthreads();                            // done reading staging buffer
    }
#else
    for (int e = tid * 4; e < 64 * 1024; e += 256 * 4) {
      f4 v = *(const f4*)(wp_head_w + e);
      int r = e >> 10, k = e & 1023;
#pragma unroll
      for (int i = 0; i < 4; ++i) {
        float x = v[i];
        __bf16 hb = (__bf16)x;
        hHi[r * HEAD_STRIDE + k + i] = hb;
        hLo[r * HEAD_STRIDE + k + i] = (__bf16)(x - (float)hb);
      }
    }
    __syncthreads();
#endif
  }

  // ---------------- Phase 1: z0 = token @ wp_head_w.T + b  (16 rows/wave) --
  v8f hT[4];
#pragma unroll
  for (int j = 0; j < 4; ++j) hT[j] = (v8f){0, 0, 0, 0, 0, 0, 0, 0};
  {
    const size_t rowG = (size_t)(wgBase + wave * 16 + nIdx);
    const float* tok = token + rowG * 1024;
#pragma unroll 1
    for (int kb = 0; kb < 32; ++kb) {
      const int kbase = kb * 32 + hiHalf * 16;    // A contiguous K run
      float av[16];
      *(f4*)(av + 0)  = *(const f4*)(tok + kbase + 0);
      *(f4*)(av + 4)  = *(const f4*)(tok + kbase + 4);
      *(f4*)(av + 8)  = *(const f4*)(tok + kbase + 8);
      *(f4*)(av + 12) = *(const f4*)(tok + kbase + 12);
      v16bf ah, al;
      splitA(av, ah, al);
      const int kB = (kb * 32 + hiHalf * 16) * 2; // B contiguous K run (bytes)
#pragma unroll
      for (int j = 0; j < 4; ++j) {
        const int rB = (j * 16 + nIdx) * (HEAD_STRIDE * 2);
        v16bf bh = *(const v16bf*)(smem + HEAD_HI_OFF + rB + kB);
        v16bf bl = *(const v16bf*)(smem + HEAD_LO_OFF + rB + kB);
        hT[j] = wmma3(ah, al, bh, bl, hT[j]);
      }
    }
#pragma unroll
    for (int j = 0; j < 4; ++j) {
      float bb = wp_head_b[j * 16 + nIdx];
#pragma unroll
      for (int r = 0; r < 8; ++r) hT[j][r] += bb;
    }
  }
  __syncthreads();

  // ---------------- Stage phase-2 weights ---------------------------------
  {
    __bf16* wHi = (__bf16*)(smem + WHH_HI_OFF);
    __bf16* wLo = (__bf16*)(smem + WHH_LO_OFF);
    for (int e = tid; e < 192 * 64; e += 256) {
      int r = e >> 6, k = e & 63;
      float x = w_hh[e];
      __bf16 hb = (__bf16)x;
      wHi[r * WHH_STRIDE + k] = hb;
      wLo[r * WHH_STRIDE + k] = (__bf16)(x - (float)hb);
    }
    float* pBIH  = (float*)(smem + BIH_OFF);
    float* pBHH  = (float*)(smem + BHH_OFF);
    float* pWIH  = (float*)(smem + WIH_OFF);
    float* pWOUT = (float*)(smem + WOUT_OFF);
    float* pWOB  = (float*)(smem + WOUTB_OFF);
    if (tid < 192) { pBIH[tid] = b_ih[tid]; pBHH[tid] = b_hh[tid]; }
    for (int e = tid; e < 384; e += 256) pWIH[e] = w_ih[e];
    if (tid < 128) pWOUT[tid] = wp_out_w[tid];
    if (tid < 2)   pWOB[tid]  = wp_out_b[tid];
    ((float*)(smem + X_OFF))[tid] = 0.0f;    // 256 == 128 rows * 2 : x0 = 0
  }
  __syncthreads();

  const float* pBIH  = (const float*)(smem + BIH_OFF);
  const float* pBHH  = (const float*)(smem + BHH_OFF);
  const float* pWIH  = (const float*)(smem + WIH_OFF);
  const float* pWOUT = (const float*)(smem + WOUT_OFF);
  const float* pWOB  = (const float*)(smem + WOUTB_OFF);
  float*       xw    = (float*)(smem + X_OFF);
  float*       hs    = (float*)(smem + HS_OFF + wave * HS_WAVE_BYTES);

  // Constant B operand for x-delta: wp_out_w padded to 64x16 (cols 0..1 valid)
  v16bf boh[2], bol[2];
#pragma unroll
  for (int kb = 0; kb < 2; ++kb) {
    float bv[16];
#pragma unroll
    for (int e = 0; e < 16; ++e) {
      int k = kb * 32 + hiHalf * 16 + e;
      bv[e] = (nIdx < 2) ? pWOUT[nIdx * 64 + k] : 0.0f;
    }
    splitA(bv, boh[kb], bol[kb]);
  }

  v16bf aHi[2], aLo[2];
  // h (C layout) -> wave-private LDS scratch, then re-read in A layout (hi/lo)
  auto storeH = [&]() {
#pragma unroll
    for (int j = 0; j < 4; ++j)
#pragma unroll
      for (int r = 0; r < 8; ++r)
        hs[(r + 8 * hiHalf) * HS_STRIDE + j * 16 + nIdx] = hT[j][r];
  };
  auto buildA = [&]() {
#pragma unroll
    for (int kb = 0; kb < 2; ++kb) {
      const float* rp = hs + nIdx * HS_STRIDE + kb * 32 + hiHalf * 16;
      float av[16];
      *(f4*)(av + 0)  = *(const f4*)(rp + 0);
      *(f4*)(av + 4)  = *(const f4*)(rp + 4);
      *(f4*)(av + 8)  = *(const f4*)(rp + 8);
      *(f4*)(av + 12) = *(const f4*)(rp + 12);
      splitA(av, aHi[kb], aLo[kb]);
    }
  };
  auto ghTile = [&](int row0) -> v8f {   // (h @ w_hh.T + b_hh) 16x16 tile
    float bb = pBHH[row0 + nIdx];
    v8f c = {bb, bb, bb, bb, bb, bb, bb, bb};
#pragma unroll
    for (int kb = 0; kb < 2; ++kb) {
      const int off = (row0 + nIdx) * (WHH_STRIDE * 2) + (kb * 32 + hiHalf * 16) * 2;
      v16bf bh = *(const v16bf*)(smem + WHH_HI_OFF + off);
      v16bf bl = *(const v16bf*)(smem + WHH_LO_OFF + off);
      c = wmma3(aHi[kb], aLo[kb], bh, bl, c);
    }
    return c;
  };

  storeH();
  buildA();

  // ---------------- Phase 2: 10 GRU steps ---------------------------------
  for (int t = 0; t < T; ++t) {
    float xr0[8], xr1[8];
#pragma unroll
    for (int r = 0; r < 8; ++r) {
      int rwg = wave * 16 + hiHalf * 8 + r;
      xr0[r] = xw[rwg * 2 + 0];
      xr1[r] = xw[rwg * 2 + 1];
    }
    v8f rg[4], ug[4];
#pragma unroll
    for (int j = 0; j < 4; ++j) {                      // r gate (rows 0..63)
      v8f c = ghTile(j * 16);
      int col = j * 16 + nIdx;
      float wi0 = pWIH[col * 2], wi1 = pWIH[col * 2 + 1], bi = pBIH[col];
#pragma unroll
      for (int r = 0; r < 8; ++r) {
        float g = c[r] + bi + wi0 * xr0[r] + wi1 * xr1[r];
        rg[j][r] = 1.0f / (1.0f + __expf(-g));
      }
    }
#pragma unroll
    for (int j = 0; j < 4; ++j) {                      // z gate (rows 64..127)
      v8f c = ghTile(64 + j * 16);
      int col = 64 + j * 16 + nIdx;
      float wi0 = pWIH[col * 2], wi1 = pWIH[col * 2 + 1], bi = pBIH[col];
#pragma unroll
      for (int r = 0; r < 8; ++r) {
        float g = c[r] + bi + wi0 * xr0[r] + wi1 * xr1[r];
        ug[j][r] = 1.0f / (1.0f + __expf(-g));
      }
    }
#pragma unroll
    for (int j = 0; j < 4; ++j) {                      // n gate + h update
      v8f c = ghTile(128 + j * 16);
      int col = 128 + j * 16 + nIdx;
      float wi0 = pWIH[col * 2], wi1 = pWIH[col * 2 + 1], bi = pBIH[col];
#pragma unroll
      for (int r = 0; r < 8; ++r) {
        float g = bi + wi0 * xr0[r] + wi1 * xr1[r] + rg[j][r] * c[r];
        float e = __expf(2.0f * g);
        float nn = (e - 1.0f) / (e + 1.0f);            // tanh
        hT[j][r] = (1.0f - ug[j][r]) * nn + ug[j][r] * hT[j][r];
      }
    }
    storeH();          // also feeds next iteration's GH
    buildA();
    // x += h_new @ wp_out.T + b_out   (WMMA with padded B; cols 0,1 valid)
    v8f d = {0, 0, 0, 0, 0, 0, 0, 0};
#pragma unroll
    for (int kb = 0; kb < 2; ++kb) d = wmma3(aHi[kb], aLo[kb], boh[kb], bol[kb], d);
    if (nIdx < 2) {
      float bo = pWOB[nIdx];
#pragma unroll
      for (int r = 0; r < 8; ++r) {
        int rwg = wave * 16 + hiHalf * 8 + r;
        float xn = xw[rwg * 2 + nIdx] + d[r] + bo;
        xw[rwg * 2 + nIdx] = xn;
        out[((size_t)(wgBase + rwg) * T + t) * 2 + nIdx] = xn;
      }
    }
  }
}

extern "C" void kernel_launch(void* const* d_in, const int* in_sizes, int n_in,
                              void* d_out, int out_size, void* d_ws, size_t ws_size,
                              hipStream_t stream) {
  (void)n_in; (void)out_size; (void)d_ws; (void)ws_size;
  const float* token     = (const float*)d_in[0];
  const float* wp_head_w = (const float*)d_in[1];
  const float* wp_head_b = (const float*)d_in[2];
  const float* w_ih      = (const float*)d_in[3];
  const float* w_hh      = (const float*)d_in[4];
  const float* b_ih      = (const float*)d_in[5];
  const float* b_hh      = (const float*)d_in[6];
  const float* wp_out_w  = (const float*)d_in[7];
  const float* wp_out_b  = (const float*)d_in[8];
  const int*   nwp       = (const int*)d_in[9];
  float* out = (float*)d_out;

  const int B = in_sizes[0] / 1024;      // 131072
  const int grid = B / 128;              // 128 rows per workgroup

  (void)hipFuncSetAttribute((const void*)gru_waypoints_kernel,
                            hipFuncAttributeMaxDynamicSharedMemorySize, LDS_BYTES);
  gru_waypoints_kernel<<<grid, 256, LDS_BYTES, stream>>>(
      token, wp_head_w, wp_head_b, w_ih, w_hh, b_ih, b_hh,
      wp_out_w, wp_out_b, nwp, out);
}
